// LearningShapeletsSeg_6416681140954
// MI455X (gfx1250) — compile-verified, matrix operand order checked by reading, not run on hardware
//
#include <hip/hip_runtime.h>
#include <hip/hip_bf16.h>
#include <math.h>

typedef __attribute__((ext_vector_type(16))) _Float16 v16h;
typedef __attribute__((ext_vector_type(8)))  _Float16 v8h;
typedef __attribute__((ext_vector_type(8)))  float    v8f;
typedef __attribute__((ext_vector_type(4)))  float    v4f;

#define T_LEN   8192
#define L_WIN   64
#define N_P     64
#define TP      (T_LEN - L_WIN + 1)   /* 8129 valid windows */
#define LEFT    (L_WIN / 2)           /* 32 */
#define EPSV    1e-6f
#define TILE_T  64
#define ASTRIDE 72                    /* padded LDS row stride (halfs) */
#define CSTRIDE 68                    /* padded LDS row stride (floats) */
#define NBATCH  32

// ---------------------------------------------------------------------------
// Kernel 1: z-normalize the 64 shapelets (C=1, L=64) -> f16 B matrix + s2[p].
// ---------------------------------------------------------------------------
__global__ __launch_bounds__(64) void shp_prep(const float* __restrict__ sh,
                                               _Float16* __restrict__ snorm,
                                               float* __restrict__ s2) {
  int p = threadIdx.x;                 // 64 shapelets, one per thread
  const float* row = sh + p * L_WIN;
  float s = 0.f, ss = 0.f;
  for (int l = 0; l < L_WIN; ++l) { float v = row[l]; s += v; ss += v * v; }
  float mean = s * (1.f / L_WIN);
  float var  = fmaxf(ss * (1.f / L_WIN) - mean * mean, 0.f);
  float sd   = sqrtf(var);
  float istd = 1.f / fmaxf(sd, EPSV);
  s2[p] = (float)L_WIN * var * istd * istd;   // sum of squares of z-normed
  for (int l = 0; l < L_WIN; ++l)
    snorm[p * L_WIN + l] = (_Float16)((row[l] - mean) * istd);
}

// ---------------------------------------------------------------------------
// Kernel 2: zero the padded time rows (t_out in [0,32) U [8161,8192)).
// ---------------------------------------------------------------------------
__global__ __launch_bounds__(256) void shp_zpad(float* __restrict__ out) {
  int i = blockIdx.x * 256 + threadIdx.x;
  const int NPAD = NBATCH * 63 * N_P;
  if (i >= NPAD) return;
  int p = i & (N_P - 1);
  int r = (i >> 6) % 63;
  int b = i / (63 * N_P);
  int t = (r < LEFT) ? r : (TP + LEFT + (r - LEFT));
  out[((size_t)b * T_LEN + t) * N_P + p] = 0.f;
}

// ---------------------------------------------------------------------------
// Kernel 3: main. One block = 64 windows x 64 shapelets for one batch.
// 4 waves; each wave: 16 rows x 4 N-tiles x K=64 -> 8x v_wmma_f32_16x16x32_f16.
// Epilogue: stage exp() tile in LDS (reusing A/B space), stream out as
// non-temporal b128 stores over one contiguous 16 KB span.
// ---------------------------------------------------------------------------
__global__ __launch_bounds__(128) void shp_main(const float* __restrict__ x,
                                                const _Float16* __restrict__ snorm,
                                                const float* __restrict__ s2g,
                                                float* __restrict__ out) {
  constexpr int AB_BYTES = (TILE_T * ASTRIDE + N_P * ASTRIDE) * (int)sizeof(_Float16); // 18432
  constexpr int C_BYTES  = TILE_T * CSTRIDE * (int)sizeof(float);                      // 17408
  constexpr int SM_BYTES = AB_BYTES > C_BYTES ? AB_BYTES : C_BYTES;

  __shared__ float xseg[TILE_T + L_WIN];                  // 128 floats
  __shared__ float meanA[TILE_T], istdA[TILE_T], w2A[TILE_T];
  __shared__ float s2l[N_P];
  __shared__ __align__(16) unsigned char smem[SM_BYTES];  // A+B, later C

  _Float16* Ald = reinterpret_cast<_Float16*>(smem);
  _Float16* Bld = Ald + TILE_T * ASTRIDE;
  float*    Cld = reinterpret_cast<float*>(smem);

  const int tid = threadIdx.x;
  const int b   = blockIdx.y;
  const int t0  = blockIdx.x * TILE_T;
  const bool fullTile = (t0 + TILE_T <= TP);              // 127 of 128 tiles
  const float* xb = x + (size_t)b * T_LEN;                // C==1

  // ---- load x segment (127 valid, clamp tail) -----------------------------
  {
    int t = t0 + tid;
    xseg[tid] = (tid < TILE_T + L_WIN - 1 && t < T_LEN) ? xb[t] : 0.f;
  }
  // ---- stage B (normalized shapelets) + s2 into LDS (dword copies) --------
  {
    const unsigned* s32 = reinterpret_cast<const unsigned*>(snorm);
    #pragma unroll
    for (int k = 0; k < 16; ++k) {
      int e   = tid + k * 128;          // dword index, 0..2047
      int row = e >> 5;                 // 32 dwords per shapelet
      int col = e & 31;
      *reinterpret_cast<unsigned*>(&Bld[row * ASTRIDE + col * 2]) = s32[e];
    }
    if (tid < N_P) s2l[tid] = s2g[tid];
  }
  __syncthreads();

  // ---- per-window mean / inv-std / w2 -------------------------------------
  if (tid < TILE_T) {
    int t = t0 + tid;
    float mean = 0.f, istd = 0.f, w2 = 0.f;
    if (t < TP) {
      float s = 0.f, ss = 0.f;
      for (int i = 0; i < L_WIN; ++i) { float v = xseg[tid + i]; s += v; ss += v * v; }
      mean = s * (1.f / L_WIN);
      float var = fmaxf(ss * (1.f / L_WIN) - mean * mean, 0.f);
      float sd  = sqrtf(var);
      istd = 1.f / fmaxf(sd, EPSV);
      w2   = (float)L_WIN * var * istd * istd;
    }
    meanA[tid] = mean; istdA[tid] = istd; w2A[tid] = w2;
  }
  __syncthreads();

  // ---- build normalized f16 A tile (64 rows x 64 k) -----------------------
  #pragma unroll
  for (int k = 0; k < 32; ++k) {
    int e   = tid + k * 128;            // 0..4095
    int row = e >> 6;
    int col = e & 63;
    float v = (xseg[row + col] - meanA[row]) * istdA[row]; // invalid rows -> 0
    Ald[row * ASTRIDE + col] = (_Float16)v;
  }
  __syncthreads();

  // ---- WMMA: D = A(16x64) x B(64x64) per wave -----------------------------
  const int wave = tid >> 5;
  const int lane = tid & 31;
  const int half = lane >> 4;           // 0: lanes 0-15, 1: lanes 16-31
  const int mrow = wave * 16 + (lane & 15);
  const int kb   = half * 8;            // per-lane K chunk base (halfs)

  v16h afrag[2];
  {
    const _Float16* ap = &Ald[mrow * ASTRIDE];
    #pragma unroll
    for (int s = 0; s < 2; ++s) {
      union { v16h v; v8h h[2]; } u;
      u.h[0] = *reinterpret_cast<const v8h*>(ap + s * 32 + kb);       // ds_load_b128
      u.h[1] = *reinterpret_cast<const v8h*>(ap + s * 32 + kb + 16);  // ds_load_b128
      afrag[s] = u.v;
    }
  }

  v8f acc[4] = {};
  #pragma unroll
  for (int nt = 0; nt < 4; ++nt) {
    const _Float16* bp = &Bld[(nt * 16 + (lane & 15)) * ASTRIDE];
    #pragma unroll
    for (int s = 0; s < 2; ++s) {
      union { v16h v; v8h h[2]; } u;
      u.h[0] = *reinterpret_cast<const v8h*>(bp + s * 32 + kb);
      u.h[1] = *reinterpret_cast<const v8h*>(bp + s * 32 + kb + 16);
      acc[nt] = __builtin_amdgcn_wmma_f32_16x16x32_f16(
          /*neg_a=*/false, afrag[s], /*neg_b=*/false, u.v,
          /*c_mod=*/(short)0, acc[nt], /*reuse_a=*/false, /*reuse_b=*/false);
    }
  }

  __syncthreads();   // everyone done reading A/B from LDS; reuse space for C

  // ---- act = exp(-(w2 + s2 - 2*dot)), staged to LDS -----------------------
  #pragma unroll
  for (int nt = 0; nt < 4; ++nt) {
    int p = nt * 16 + (lane & 15);
    float s2p = s2l[p];
    #pragma unroll
    for (int r = 0; r < 8; ++r) {
      int m = wave * 16 + half * 8 + r;   // C/D layout: VGPR r -> M=r or r+8
      float d2 = w2A[m] + s2p - 2.f * acc[nt][r];
      Cld[m * CSTRIDE + p] = __expf(-d2);
    }
  }
  __syncthreads();

  // ---- stream the 64x64 tile out: contiguous 16 KB, NT b128 stores --------
  float* ob = out + ((size_t)b * T_LEN + (t0 + LEFT)) * N_P;  // row stride == 64
  if (fullTile) {
    #pragma unroll
    for (int k = 0; k < 8; ++k) {
      int c   = tid + k * 128;            // float4 chunk, 0..1023
      int row = c >> 4;
      v4f v = *reinterpret_cast<const v4f*>(&Cld[row * CSTRIDE + (c & 15) * 4]);
      __builtin_nontemporal_store(v, reinterpret_cast<v4f*>(ob + c * 4));
    }
  } else {
    #pragma unroll
    for (int k = 0; k < 8; ++k) {
      int c   = tid + k * 128;
      int row = c >> 4;
      if (t0 + row < TP) {
        v4f v = *reinterpret_cast<const v4f*>(&Cld[row * CSTRIDE + (c & 15) * 4]);
        __builtin_nontemporal_store(v, reinterpret_cast<v4f*>(ob + c * 4));
      }
    }
  }
}

// ---------------------------------------------------------------------------
extern "C" void kernel_launch(void* const* d_in, const int* in_sizes, int n_in,
                              void* d_out, int out_size, void* d_ws, size_t ws_size,
                              hipStream_t stream) {
  const float* x  = (const float*)d_in[0];   // (32, 8192, 1) fp32
  const float* sh = (const float*)d_in[1];   // (64, 1, 64)  fp32
  float* out = (float*)d_out;                // (32, 8192, 64) fp32

  _Float16* snorm = (_Float16*)d_ws;                                   // 8 KB
  float*    s2    = (float*)((char*)d_ws + N_P * L_WIN * sizeof(_Float16));

  shp_prep<<<1, 64, 0, stream>>>(sh, snorm, s2);

  const int npad = NBATCH * 63 * N_P;
  shp_zpad<<<(npad + 255) / 256, 256, 0, stream>>>(out);

  dim3 grid((TP + TILE_T - 1) / TILE_T, NBATCH);   // (128, 32)
  shp_main<<<grid, 128, 0, stream>>>(x, snorm, s2, out);
}
